// GraphBasedMemBank_9225589752442
// MI455X (gfx1250) — matrix operand order repaired and checked by reading, not compiled, not run on hardware
//
#include <hip/hip_runtime.h>
#include <math.h>

typedef __bf16 bf16_t;
typedef __attribute__((ext_vector_type(16))) __bf16 v16bf;
typedef __attribute__((ext_vector_type(8)))  __bf16 v8bf;
typedef __attribute__((ext_vector_type(8)))  float   v8f;

#define BB 2
#define TT 8
#define MM 512
#define DD 1024
#define HH 1024
#define WWIN 3
#define KNN 8
#define BM (BB*MM)

// ---------------------------------------------------------------------------
// Generic NT GEMM: C[m][n] = sum_k A[m][k] * X[n][k]   (bf16 in, f32 acc)
// Workgroup: 256 threads = 8 waves; tile 64(M) x 128(N); wave tile 16x64.
// A frag (16x32 bf16): lane L -> row mBase+(L%16); elem e: A[m][k+16g+e]
//   (g = L/16)   -> one contiguous 32B load.
// B frag (32x16 bf16): lane L -> col nBase+(L%16); elem e: X[n][k+16g+e]
//   -> one contiguous 32B load (NT layout: rows of X).
// C/D: lane L -> n = nBase+(L%16); vgpr i -> m = mBase + 8g + i.
// ---------------------------------------------------------------------------
__global__ __launch_bounds__(256)
void gemm_nt_wmma(const bf16_t* __restrict__ A, int lda, long sA,
                  const bf16_t* __restrict__ X, int ldx, long sX,
                  float* __restrict__ C, int ldc, long sC,
                  const float* __restrict__ bias,
                  const float* __restrict__ E, int lde, long sE,
                  bf16_t* __restrict__ Obf, int ldo, long sO,
                  int Kdim, int addC)
{
    const int lane = threadIdx.x & 31;
    const int wave = threadIdx.x >> 5;
    const int wm   = wave & 3;            // 4 waves along M
    const int wn   = wave >> 2;           // 2 waves along N
    const int mBase = blockIdx.y * 64 + wm * 16;
    const int nBase = blockIdx.x * 128 + wn * 64;
    const long zb = blockIdx.z;

    A += zb * sA; X += zb * sX; C += zb * sC;
    if (E)   E   += zb * sE;
    if (Obf) Obf += zb * sO;

    const int g   = lane >> 4;
    const int l16 = lane & 15;

    const bf16_t* arow = A + (long)(mBase + l16) * lda;
    const bf16_t* x0 = X + (long)(nBase + l16 +  0) * ldx;
    const bf16_t* x1 = X + (long)(nBase + l16 + 16) * ldx;
    const bf16_t* x2 = X + (long)(nBase + l16 + 32) * ldx;
    const bf16_t* x3 = X + (long)(nBase + l16 + 48) * ldx;

    v8f acc[4];
    acc[0] = (v8f){}; acc[1] = (v8f){}; acc[2] = (v8f){}; acc[3] = (v8f){};

    for (int k = 0; k < Kdim; k += 32) {
        v16bf a = *(const v16bf*)(arow + k + 16 * g);
        v16bf b0 = *(const v16bf*)(x0 + k + 16 * g);
        v16bf b1 = *(const v16bf*)(x1 + k + 16 * g);
        v16bf b2 = *(const v16bf*)(x2 + k + 16 * g);
        v16bf b3 = *(const v16bf*)(x3 + k + 16 * g);
        acc[0] = __builtin_amdgcn_wmma_f32_16x16x32_bf16(false, a, false, b0, (short)0, acc[0], false, false);
        acc[1] = __builtin_amdgcn_wmma_f32_16x16x32_bf16(false, a, false, b1, (short)0, acc[1], false, false);
        acc[2] = __builtin_amdgcn_wmma_f32_16x16x32_bf16(false, a, false, b2, (short)0, acc[2], false, false);
        acc[3] = __builtin_amdgcn_wmma_f32_16x16x32_bf16(false, a, false, b3, (short)0, acc[3], false, false);
    }

    const int mRow = mBase + 8 * g;
#pragma unroll
    for (int nt = 0; nt < 4; ++nt) {
        const int n = nBase + nt * 16 + l16;
        float bv = bias ? bias[n] : 0.0f;
#pragma unroll
        for (int i = 0; i < 8; ++i) {
            const long m = mRow + i;
            float v = acc[nt][i] + bv;
            long off = m * (long)ldc + n;
            if (addC) v += C[off];
            if (E)    v += E[m * (long)lde + n];
            C[off] = v;
            if (Obf) Obf[m * (long)ldo + n] = (bf16_t)v;
        }
    }
}

// --------------------------- elementwise kernels ---------------------------
__global__ __launch_bounds__(256)
void f32_to_bf16_k(const float* __restrict__ src, bf16_t* __restrict__ dst, long n)
{
    for (long i = blockIdx.x * 256L + threadIdx.x; i < n; i += (long)gridDim.x * 256L)
        dst[i] = (bf16_t)src[i];
}

// one block per row: write z_bf and zn_bf (= z / max(||z||, 1e-12))
__global__ __launch_bounds__(256)
void normalize_rows_k(const float* __restrict__ z, bf16_t* __restrict__ z_bf,
                      bf16_t* __restrict__ zn_bf)
{
    const long row = blockIdx.x;
    const float* p = z + row * DD;
    __shared__ float red[256];
    float ss = 0.0f;
    for (int d = threadIdx.x; d < DD; d += 256) { float v = p[d]; ss += v * v; }
    red[threadIdx.x] = ss; __syncthreads();
    for (int o = 128; o > 0; o >>= 1) {
        if (threadIdx.x < o) red[threadIdx.x] += red[threadIdx.x + o];
        __syncthreads();
    }
    float inv = 1.0f / fmaxf(sqrtf(red[0]), 1e-12f);
    for (int d = threadIdx.x; d < DD; d += 256) {
        float v = p[d];
        z_bf [row * DD + d] = (bf16_t)v;
        zn_bf[row * DD + d] = (bf16_t)(v * inv);
    }
}

__global__ __launch_bounds__(256)
void wksum_k(const float* __restrict__ Wk, float* __restrict__ wksum)
{
    int d = blockIdx.x * 256 + threadIdx.x;
    if (d < DD) {
        float s = 0.0f;
        for (int i = 0; i < DD; ++i) s += Wk[(long)i * DD + d];
        wksum[d] = s;
    }
}

__global__ __launch_bounds__(256)
void s0_k(const float* __restrict__ z, const float* __restrict__ Wq_att,
          float* __restrict__ s0, int t)
{
    const int b = blockIdx.x;
    const float* zr = z + (((long)b * TT + t) * MM + 0) * DD;
    __shared__ float red[256];
    float s = 0.0f;
    for (int d = threadIdx.x; d < DD; d += 256) s += zr[d] * Wq_att[d]; // row 0 of Watt_q
    red[threadIdx.x] = s; __syncthreads();
    for (int o = 128; o > 0; o >>= 1) {
        if (threadIdx.x < o) red[threadIdx.x] += red[threadIdx.x + o];
        __syncthreads();
    }
    if (threadIdx.x == 0) s0[b] = red[0];
}

__global__ __launch_bounds__(256)
void zero_mem_k(float* __restrict__ mem, bf16_t* __restrict__ mem_bf, long n)
{
    long i = blockIdx.x * 256L + threadIdx.x;
    if (i < n) { mem[i] = 0.0f; mem_bf[i] = (bf16_t)0.0f; }
}

// top-8 (desc, ties -> lower index): one thread per (b,m) row, contiguous valid cols
__global__ __launch_bounds__(256)
void topk_k(const float* __restrict__ sim, float* __restrict__ vals,
            int* __restrict__ idx, int ncols)
{
    int r = blockIdx.x * 256 + threadIdx.x;
    if (r >= BM) return;
    const float* row = sim + (long)r * (WWIN * MM);
    float tv[KNN]; int ti[KNN];
#pragma unroll
    for (int k = 0; k < KNN; ++k) { tv[k] = -__builtin_huge_valf(); ti[k] = 0; }
    for (int c = 0; c < ncols; ++c) {
        float v = row[c];
        if (v > tv[KNN - 1]) {
            int j = KNN - 1;
            while (j > 0 && v > tv[j - 1]) { tv[j] = tv[j - 1]; ti[j] = ti[j - 1]; --j; }
            tv[j] = v; ti[j] = c;
        }
    }
#pragma unroll
    for (int k = 0; k < KNN; ++k) { vals[(long)r * KNN + k] = tv[k]; idx[(long)r * KNN + k] = ti[k]; }
}

// per (b,m): kh_sum dots, softmax att, alpha-weighted gather -> ctx_bf
__global__ __launch_bounds__(256)
void ctx_k(const float* __restrict__ z, const float* __restrict__ wksum,
           const float* __restrict__ s0v, const float* __restrict__ vals,
           const int* __restrict__ idx, bf16_t* __restrict__ ctx_bf, int t)
{
    const int r = blockIdx.x;
    const int b = r / MM;
    __shared__ float red[256];
    __shared__ float kh[KNN];
    __shared__ float alpha[KNN];
    __shared__ long  rowoff[KNN];
    if (threadIdx.x < KNN) {
        int k = threadIdx.x;
        int id = idx[(long)r * KNN + k];
        int dt = id / MM, j = id % MM;
        int tt = t - dt;
        rowoff[k] = (((long)b * TT + tt) * MM + j) * DD;
    }
    __syncthreads();
    for (int k = 0; k < KNN; ++k) {
        const float* zr = z + rowoff[k];
        float s = 0.0f;
        for (int d = threadIdx.x; d < DD; d += 256) s += zr[d] * wksum[d];
        red[threadIdx.x] = s; __syncthreads();
        for (int o = 128; o > 0; o >>= 1) {
            if (threadIdx.x < o) red[threadIdx.x] += red[threadIdx.x + o];
            __syncthreads();
        }
        if (threadIdx.x == 0) kh[k] = red[0];
        __syncthreads();
    }
    if (threadIdx.x == 0) {
        float s0 = s0v[b];
        float mx = -__builtin_huge_valf();
        for (int k = 0; k < KNN; ++k) mx = fmaxf(mx, s0 * kh[k]);
        float e[KNN], sum = 0.0f;
        for (int k = 0; k < KNN; ++k) { e[k] = expf(s0 * kh[k] - mx); sum += e[k]; }
        for (int k = 0; k < KNN; ++k)
            alpha[k] = 0.5f * vals[(long)r * KNN + k] + 0.5f * e[k] / sum;
    }
    __syncthreads();
    for (int d = threadIdx.x; d < DD; d += 256) {
        float acc = 0.0f;
#pragma unroll
        for (int k = 0; k < KNN; ++k) acc += alpha[k] * z[rowoff[k] + d];
        ctx_bf[(long)r * DD + d] = (bf16_t)acc;
    }
}

// layernorm of xbuf row -> xn_bf
__global__ __launch_bounds__(256)
void ln_k(const float* __restrict__ x, const float* __restrict__ w,
          const float* __restrict__ bns, bf16_t* __restrict__ out)
{
    const long r = blockIdx.x;
    const float* p = x + r * DD;
    __shared__ float red[256];
    __shared__ float mu_s, inv_s;
    float s = 0.0f;
    for (int d = threadIdx.x; d < DD; d += 256) s += p[d];
    red[threadIdx.x] = s; __syncthreads();
    for (int o = 128; o > 0; o >>= 1) {
        if (threadIdx.x < o) red[threadIdx.x] += red[threadIdx.x + o];
        __syncthreads();
    }
    if (threadIdx.x == 0) mu_s = red[0] / DD;
    __syncthreads();
    float mu = mu_s;
    float v = 0.0f;
    for (int d = threadIdx.x; d < DD; d += 256) { float t = p[d] - mu; v += t * t; }
    red[threadIdx.x] = v; __syncthreads();
    for (int o = 128; o > 0; o >>= 1) {
        if (threadIdx.x < o) red[threadIdx.x] += red[threadIdx.x + o];
        __syncthreads();
    }
    if (threadIdx.x == 0) inv_s = rsqrtf(red[0] / DD + 1e-5f);
    __syncthreads();
    float inv = inv_s;
    for (int d = threadIdx.x; d < DD; d += 256)
        out[r * DD + d] = (bf16_t)(((p[d] - mu) * inv) * w[d] + bns[d]);
}

// GRU gate fusion: mem = (1-zg)*n + zg*mem
__global__ __launch_bounds__(256)
void gru_k(const float* __restrict__ gi, const float* __restrict__ gh,
           float* __restrict__ mem, bf16_t* __restrict__ mem_bf)
{
    long e = blockIdx.x * 256L + threadIdx.x;
    if (e >= (long)BM * HH) return;
    long r = e / HH, h = e % HH;
    long base = r * (3L * HH);
    float ir  = gi[base + h],           hr = gh[base + h];
    float iz  = gi[base + HH + h],      hz = gh[base + HH + h];
    float inn = gi[base + 2 * HH + h],  hn = gh[base + 2 * HH + h];
    float rg = 1.0f / (1.0f + expf(-(ir + hr)));
    float zg = 1.0f / (1.0f + expf(-(iz + hz)));
    float nn = tanhf(inn + rg * hn);
    float m  = mem[e];
    m = (1.0f - zg) * nn + zg * m;
    mem[e] = m;
    mem_bf[e] = (bf16_t)m;
}

// ---------------------------------------------------------------------------
extern "C" void kernel_launch(void* const* d_in, const int* in_sizes, int n_in,
                              void* d_out, int out_size, void* d_ws, size_t ws_size,
                              hipStream_t stream)
{
    (void)in_sizes; (void)n_in; (void)out_size; (void)ws_size;
    const float* z      = (const float*)d_in[0];
    const float* Wq     = (const float*)d_in[1];
    const float* Wv     = (const float*)d_in[2];
    const float* Wout   = (const float*)d_in[3];
    const float* ln_w   = (const float*)d_in[4];
    const float* ln_b   = (const float*)d_in[5];
    const float* Watt_q = (const float*)d_in[6];
    const float* Watt_k = (const float*)d_in[7];
    const float* W_ih   = (const float*)d_in[8];
    const float* W_hh   = (const float*)d_in[9];
    const float* b_ih   = (const float*)d_in[10];
    const float* b_hh   = (const float*)d_in[11];
    const float* W_mf   = (const float*)d_in[12];
    const float* b_mf   = (const float*)d_in[13];

    // ---- carve workspace ----
    size_t off = 0;
    auto alloc = [&](size_t bytes) -> void* {
        void* p = (char*)d_ws + off;
        off = (off + bytes + 255) & ~(size_t)255;
        return p;
    };
    const long ZN = (long)BB * TT * MM * DD;
    bf16_t* z_bf    = (bf16_t*)alloc(ZN * 2);
    bf16_t* zn_bf   = (bf16_t*)alloc(ZN * 2);
    bf16_t* Wq_bf   = (bf16_t*)alloc((long)DD * DD * 2);
    bf16_t* Wv_bf   = (bf16_t*)alloc((long)DD * DD * 2);
    bf16_t* Wout_bf = (bf16_t*)alloc((long)DD * DD * 2);
    bf16_t* Wih_bf  = (bf16_t*)alloc(3L * HH * DD * 2);
    bf16_t* Whh_bf  = (bf16_t*)alloc(3L * HH * HH * 2);
    bf16_t* Wmf_bf  = (bf16_t*)alloc((long)DD * HH * 2);
    float*  wksum   = (float*)alloc(DD * 4);
    float*  s0      = (float*)alloc(BB * 4);
    float*  sim     = (float*)alloc((long)BM * WWIN * MM * 4);
    float*  vals    = (float*)alloc((long)BM * KNN * 4);
    int*    idx     = (int*)  alloc((long)BM * KNN * 4);
    bf16_t* ctx_bf  = (bf16_t*)alloc((long)BM * DD * 2);
    float*  xbuf    = (float*)alloc((long)BM * DD * 4);
    bf16_t* xn_bf   = (bf16_t*)alloc((long)BM * DD * 2);
    float*  inpf    = (float*)alloc((long)BM * DD * 4);
    bf16_t* inp_bf  = (bf16_t*)alloc((long)BM * DD * 2);
    float*  gi      = (float*)alloc((long)BM * 3 * HH * 4);
    float*  gh      = (float*)alloc((long)BM * 3 * HH * 4);
    float*  mem     = (float*)alloc((long)BM * HH * 4);
    bf16_t* mem_bf  = (bf16_t*)alloc((long)BM * HH * 2);

    float* out_h   = (float*)d_out;                 // (B,T,M,D)
    float* out_mem = out_h + ZN;                    // (B,M,H)

    // ---- prep ----
    auto conv = [&](const float* s, bf16_t* d, long n) {
        int blocks = (int)((n + 255) / 256); if (blocks > 8192) blocks = 8192;
        f32_to_bf16_k<<<blocks, 256, 0, stream>>>(s, d, n);
    };
    conv(Wq, Wq_bf, (long)DD * DD);
    conv(Wv, Wv_bf, (long)DD * DD);
    conv(Wout, Wout_bf, (long)DD * DD);
    conv(W_ih, Wih_bf, 3L * HH * DD);
    conv(W_hh, Whh_bf, 3L * HH * HH);
    conv(W_mf, Wmf_bf, (long)DD * HH);
    normalize_rows_k<<<BB * TT * MM, 256, 0, stream>>>(z, z_bf, zn_bf);
    wksum_k<<<(DD + 255) / 256, 256, 0, stream>>>(Watt_k, wksum);
    zero_mem_k<<<(int)(((long)BM * HH + 255) / 256), 256, 0, stream>>>(mem, mem_bf, (long)BM * HH);

    auto gemm = [&](const bf16_t* A, int lda, long sA,
                    const bf16_t* X, int ldx, long sX,
                    float* C, int ldc, long sC, const float* bias,
                    const float* E, int lde, long sE,
                    bf16_t* O, int ldo, long sO,
                    int Mrows, int N, int Kd, int addC, int batches) {
        dim3 grid(N / 128, Mrows / 64, batches);
        gemm_nt_wmma<<<grid, 256, 0, stream>>>(A, lda, sA, X, ldx, sX, C, ldc, sC,
                                               bias, E, lde, sE, O, ldo, sO, Kd, addC);
    };

    const long ZTS = (long)TT * MM * DD;   // batch stride within z-shaped tensors
    const long MD  = (long)MM * DD;
    const long MH  = (long)MM * HH;

    for (int t = 0; t < TT; ++t) {
        const int nw = (t + 1 < WWIN) ? (t + 1) : WWIN;
        // 1) similarity GEMMs: sim[b][m][w*M+n] = zn_t[m] . zn_{t-w}[n]
        for (int b = 0; b < BB; ++b)
            for (int w = 0; w < nw; ++w) {
                const bf16_t* At = zn_bf + ((long)b * TT + t) * MD;
                const bf16_t* Xt = zn_bf + ((long)b * TT + (t - w)) * MD;
                float* Ct = sim + (long)b * MM * (WWIN * MM) + (long)w * MM;
                gemm(At, DD, 0, Xt, DD, 0, Ct, WWIN * MM, 0,
                     nullptr, nullptr, 0, 0, nullptr, 0, 0,
                     MM, MM, DD, 0, 1);
            }
        // 2) top-8
        topk_k<<<(BM + 255) / 256, 256, 0, stream>>>(sim, vals, idx, nw * MM);
        // 3) s0 scalar per batch
        s0_k<<<BB, 256, 0, stream>>>(z, Watt_q, s0, t);
        // 4) context = sum_k alpha_k * z[gather]
        ctx_k<<<BM, 256, 0, stream>>>(z, wksum, s0, vals, idx, ctx_bf, t);
        // 5) x = z_t @ Wq.T
        gemm(z_bf + (long)t * MD, DD, ZTS, Wq_bf, DD, 0, xbuf, DD, MD,
             nullptr, nullptr, 0, 0, nullptr, 0, 0, MM, DD, DD, 0, BB);
        // 6) x += ctx @ Wv.T
        gemm(ctx_bf, DD, MD, Wv_bf, DD, 0, xbuf, DD, MD,
             nullptr, nullptr, 0, 0, nullptr, 0, 0, MM, DD, DD, 1, BB);
        // 7) layernorm -> xn_bf
        ln_k<<<BM, 256, 0, stream>>>(xbuf, ln_w, ln_b, xn_bf);
        // 8) inp = xn @ Wout.T
        gemm(xn_bf, DD, MD, Wout_bf, DD, 0, inpf, DD, MD,
             nullptr, nullptr, 0, 0, inp_bf, DD, MD, MM, DD, DD, 0, BB);
        // 9) gi = inp @ W_ih.T + b_ih
        gemm(inp_bf, DD, MD, Wih_bf, DD, 0, gi, 3 * HH, (long)MM * 3 * HH,
             b_ih, nullptr, 0, 0, nullptr, 0, 0, MM, 3 * HH, DD, 0, BB);
        // 10) gh = mem @ W_hh.T + b_hh
        gemm(mem_bf, HH, MH, Whh_bf, HH, 0, gh, 3 * HH, (long)MM * 3 * HH,
             b_hh, nullptr, 0, 0, nullptr, 0, 0, MM, 3 * HH, HH, 0, BB);
        // 11) GRU update
        gru_k<<<(int)(((long)BM * HH + 255) / 256), 256, 0, stream>>>(gi, gh, mem, mem_bf);
        // 12) h_out = z_t + mem @ W_mf.T + b_mf  -> write directly into d_out
        gemm(mem_bf, HH, MH, Wmf_bf, HH, 0, out_h + (long)t * MD, DD, ZTS,
             b_mf, z + (long)t * MD, DD, ZTS, nullptr, 0, 0, MM, DD, HH, 0, BB);
    }

    // final mem -> tail of d_out
    hipMemcpyAsync(out_mem, mem, (long)BM * HH * sizeof(float),
                   hipMemcpyDeviceToDevice, stream);
}
